// GPSEncoder_82265803588281
// MI455X (gfx1250) — matrix operand order repaired and checked by reading, not compiled, not run on hardware
//
#include <hip/hip_runtime.h>
#include <hip/hip_bf16.h>

typedef __attribute__((ext_vector_type(16))) _Float16 v16h;
typedef __attribute__((ext_vector_type(8)))  float    v8f;

#define GN   4096
#define GE   131072
#define GHD  256
#define GNPD 192
#define GPID 20

// ======================= WMMA GEMM: C = act(A@W + bias) [+ res] =======================
// A: [M,K] f32 (lda), W: [K,Nn] f32 (ldw), C: [M,Nn] f32 (ldc).
// Block tile 128x64, 4 waves x 128 threads; wave w owns rows [w*32, w*32+32), all 64 cols.
// Per K-step each wave: 2 A-fragments + 4 B-fragments -> 8 v_wmma.
template<int ACT>
__global__ __launch_bounds__(128)
void gemm_bias_act(const float* __restrict__ A, int lda,
                   const float* __restrict__ W, int ldw,
                   const float* __restrict__ bias,
                   const float* __restrict__ res, int ldr,
                   float* __restrict__ C, int ldc,
                   int M, int Nn, int K)
{
    __shared__ _Float16 As[128 * 40];  // [row][k], row stride 40 halves (80B, 16B aligned)
    __shared__ _Float16 Bt[64 * 40];   // [n][k]   (W tile transposed)

    const int tid   = threadIdx.x;
    const int wave  = tid >> 5;
    const int lane  = tid & 31;
    const int lhalf = lane >> 4;   // 0/1
    const int lmod  = lane & 15;
    const int m0 = blockIdx.y * 128;
    const int n0 = blockIdx.x * 64;

    v8f acc[2][4] = {};

    for (int k0 = 0; k0 < K; k0 += 32) {
        __syncthreads();
        // ---- stage A tile 128x32 (one full row of 32 k per thread)
        {
            const int r = tid;
            const float* src = A + (long)(m0 + r) * lda + k0;
            const bool rok = (m0 + r) < M;
#pragma unroll
            for (int i = 0; i < 32; ++i) {
                float v = 0.f;
                if (rok && (k0 + i) < K) v = src[i];
                As[r * 40 + i] = (_Float16)v;
            }
        }
        // ---- stage W tile 32x64 transposed: Bt[n][k] = W[k0+k][n0+n]
        {
            const int kk = tid >> 2;          // 0..31
            const int nn = (tid & 3) * 16;    // 0,16,32,48
            const float* src = W + (long)(k0 + kk) * ldw + (n0 + nn);
            const bool kok = (k0 + kk) < K;
#pragma unroll
            for (int i = 0; i < 16; ++i) {
                float v = 0.f;
                if (kok && (n0 + nn + i) < Nn) v = src[i];
                Bt[(nn + i) * 40 + kk] = (_Float16)v;
            }
        }
        // ---- prefetch next K-tiles (gfx1250 global_prefetch_b8)
        if (k0 + 32 < K) {
            if ((m0 + tid) < M)
                __builtin_prefetch(A + (long)(m0 + tid) * lda + k0 + 32, 0, 1);
            const int kk = tid >> 2, nn = (tid & 3) * 16;
            if ((k0 + 32 + kk) < K)
                __builtin_prefetch(W + (long)(k0 + 32 + kk) * ldw + (n0 + nn), 0, 1);
        }
        __syncthreads();

        // ---- A fragments (two 16-row tiles per wave)
        v16h af0, af1;
        {
            const _Float16* p0 = &As[(wave * 32 + lmod) * 40 + lhalf * 8];
            const _Float16* p1 = &As[(wave * 32 + 16 + lmod) * 40 + lhalf * 8];
#pragma unroll
            for (int i = 0; i < 8; ++i) {
                af0[i] = p0[i]; af0[8 + i] = p0[16 + i];
                af1[i] = p1[i]; af1[8 + i] = p1[16 + i];
            }
        }
        // ---- 4 x (B fragment + 2 WMMA)
#pragma unroll
        for (int nb = 0; nb < 4; ++nb) {
            v16h bf;
            const _Float16* p = &Bt[(nb * 16 + lmod) * 40 + lhalf * 8];
#pragma unroll
            for (int i = 0; i < 8; ++i) { bf[i] = p[i]; bf[8 + i] = p[16 + i]; }
            acc[0][nb] = __builtin_amdgcn_wmma_f32_16x16x32_f16(
                false, af0, false, bf, (short)0, acc[0][nb], false, false);
            acc[1][nb] = __builtin_amdgcn_wmma_f32_16x16x32_f16(
                false, af1, false, bf, (short)0, acc[1][nb], false, false);
        }
    }

    // ---- epilogue: bias, activation, residual, store
#pragma unroll
    for (int sub = 0; sub < 2; ++sub) {
#pragma unroll
        for (int nb = 0; nb < 4; ++nb) {
            const int n = n0 + nb * 16 + lmod;
            if (n >= Nn) continue;
            const float bv = bias ? bias[n] : 0.f;
#pragma unroll
            for (int j = 0; j < 8; ++j) {
                const int m = m0 + wave * 32 + sub * 16 + lhalf * 8 + j;
                if (m >= M) continue;
                float v = acc[sub][nb][j] + bv;
                if (ACT == 1) v = v > 0.f ? v : 0.f;
                if (res) v += res[(long)m * ldr + n];
                C[(long)m * ldc + n] = v;
            }
        }
    }
}

// ======================= Flash attention (NH=4, dh=64, N=4096) =======================
// grid: (N/64, NH); block: 128 (4 waves). Q pre-scaled by 1/sqrt(64).
__global__ __launch_bounds__(128)
void flash_attn(const float* __restrict__ Qg, const float* __restrict__ Kg,
                const float* __restrict__ Vg, float* __restrict__ Og)
{
    __shared__ _Float16 Qs[64 * 72];        // [q][d]
    __shared__ _Float16 Ks[64 * 72];        // [j][d]  (B-operand of Q@K^T: n=j, k=d)
    __shared__ _Float16 Vt[64 * 72];        // [d][j]  (B-operand of P@V:   n=d, k=j)
    __shared__ _Float16 Ps[64 * 72];        // per-wave 16x64 P tiles

    const int tid   = threadIdx.x;
    const int wave  = tid >> 5;
    const int lane  = tid & 31;
    const int lhalf = lane >> 4;
    const int lmod  = lane & 15;
    const int head  = blockIdx.y;
    const int q0    = blockIdx.x * 64;
    const float scale = 0.125f;             // 1/sqrt(64)

    // load + scale Q tile
    {
        const int row = tid >> 1, d0 = (tid & 1) * 32;
        const float* src = Qg + (long)(q0 + row) * GHD + head * 64 + d0;
#pragma unroll
        for (int i = 0; i < 32; ++i) Qs[row * 72 + d0 + i] = (_Float16)(src[i] * scale);
    }

    float mstate[8], lstate[8];
    v8f oacc[4] = {};
#pragma unroll
    for (int j = 0; j < 8; ++j) { mstate[j] = -1e30f; lstate[j] = 0.f; }

    for (int kb = 0; kb < GN / 64; ++kb) {
        __syncthreads();   // previous iteration's P@V reads done
        {
            const int row = tid >> 1, d0 = (tid & 1) * 32;
            const float* ksrc = Kg + (long)(kb * 64 + row) * GHD + head * 64 + d0;
            const float* vsrc = Vg + (long)(kb * 64 + row) * GHD + head * 64 + d0;
#pragma unroll
            for (int i = 0; i < 32; ++i) {
                Ks[row * 72 + d0 + i]   = (_Float16)ksrc[i];
                Vt[(d0 + i) * 72 + row] = (_Float16)vsrc[i];
            }
            // prefetch next key/value tile while we do WMMA work on this one
            if (kb + 1 < GN / 64) {
                __builtin_prefetch(Kg + (long)((kb + 1) * 64 + row) * GHD + head * 64 + d0, 0, 1);
                __builtin_prefetch(Vg + (long)((kb + 1) * 64 + row) * GHD + head * 64 + d0, 0, 1);
            }
        }
        __syncthreads();

        // S = Q @ K^T (contraction over d: two k-steps of 32)
        v8f sacc[4] = {};
#pragma unroll
        for (int ks = 0; ks < 64; ks += 32) {
            v16h af;
            {
                const _Float16* p = &Qs[(wave * 16 + lmod) * 72 + ks + lhalf * 8];
#pragma unroll
                for (int i = 0; i < 8; ++i) { af[i] = p[i]; af[8 + i] = p[16 + i]; }
            }
#pragma unroll
            for (int nb = 0; nb < 4; ++nb) {
                v16h bf;
                const _Float16* p = &Ks[(nb * 16 + lmod) * 72 + ks + lhalf * 8];
#pragma unroll
                for (int i = 0; i < 8; ++i) { bf[i] = p[i]; bf[8 + i] = p[16 + i]; }
                sacc[nb] = __builtin_amdgcn_wmma_f32_16x16x32_f16(
                    false, af, false, bf, (short)0, sacc[nb], false, false);
            }
        }

        // online softmax: rows live on 16-lane half-groups
#pragma unroll
        for (int j = 0; j < 8; ++j) {
            float rm = fmaxf(fmaxf(sacc[0][j], sacc[1][j]), fmaxf(sacc[2][j], sacc[3][j]));
            rm = fmaxf(rm, __shfl_xor(rm, 1, 32));
            rm = fmaxf(rm, __shfl_xor(rm, 2, 32));
            rm = fmaxf(rm, __shfl_xor(rm, 4, 32));
            rm = fmaxf(rm, __shfl_xor(rm, 8, 32));
            const float mn = fmaxf(mstate[j], rm);
            const float alpha = __expf(mstate[j] - mn);
            mstate[j] = mn;
            float rs = 0.f;
#pragma unroll
            for (int nb = 0; nb < 4; ++nb) {
                const float pv = __expf(sacc[nb][j] - mn);
                sacc[nb][j] = pv;
                rs += pv;
            }
            rs += __shfl_xor(rs, 1, 32);
            rs += __shfl_xor(rs, 2, 32);
            rs += __shfl_xor(rs, 4, 32);
            rs += __shfl_xor(rs, 8, 32);
            lstate[j] = lstate[j] * alpha + rs;
#pragma unroll
            for (int nb = 0; nb < 4; ++nb) oacc[nb][j] *= alpha;
        }

        // spill P (C-layout) to LDS so it can be re-fragmented as a WMMA A-operand
#pragma unroll
        for (int nb = 0; nb < 4; ++nb)
#pragma unroll
            for (int j = 0; j < 8; ++j)
                Ps[(wave * 16 + lhalf * 8 + j) * 72 + nb * 16 + lmod] = (_Float16)sacc[nb][j];
        __syncthreads();

        // O += P @ V (contraction over keys: two k-steps of 32)
#pragma unroll
        for (int ks = 0; ks < 64; ks += 32) {
            v16h af;
            {
                const _Float16* p = &Ps[(wave * 16 + lmod) * 72 + ks + lhalf * 8];
#pragma unroll
                for (int i = 0; i < 8; ++i) { af[i] = p[i]; af[8 + i] = p[16 + i]; }
            }
#pragma unroll
            for (int nb = 0; nb < 4; ++nb) {
                v16h bf;
                const _Float16* p = &Vt[(nb * 16 + lmod) * 72 + ks + lhalf * 8];
#pragma unroll
                for (int i = 0; i < 8; ++i) { bf[i] = p[i]; bf[8 + i] = p[16 + i]; }
                oacc[nb] = __builtin_amdgcn_wmma_f32_16x16x32_f16(
                    false, af, false, bf, (short)0, oacc[nb], false, false);
            }
        }
    }

    // write O / l
#pragma unroll
    for (int nb = 0; nb < 4; ++nb) {
        const int d = head * 64 + nb * 16 + lmod;
#pragma unroll
        for (int j = 0; j < 8; ++j) {
            const int q = q0 + wave * 16 + lhalf * 8 + j;
            Og[(long)q * GHD + d] = oacc[nb][j] / lstate[j];
        }
    }
}

// ======================= small support kernels =======================
__global__ void bn_stats(const float* __restrict__ X, int R, int C, float* __restrict__ stats)
{
    const int c = blockIdx.x;
    float s1 = 0.f, s2 = 0.f;
    for (int r = threadIdx.x; r < R; r += 256) {
        const float v = X[(long)r * C + c];
        s1 += v; s2 += v * v;
    }
    __shared__ float sh1[256], sh2[256];
    sh1[threadIdx.x] = s1; sh2[threadIdx.x] = s2;
    __syncthreads();
    for (int s = 128; s > 0; s >>= 1) {
        if (threadIdx.x < s) { sh1[threadIdx.x] += sh1[threadIdx.x + s]; sh2[threadIdx.x] += sh2[threadIdx.x + s]; }
        __syncthreads();
    }
    if (threadIdx.x == 0) {
        const float mean = sh1[0] / (float)R;
        const float var  = sh2[0] / (float)R - mean * mean;   // biased, like jnp.var
        stats[c]     = mean;
        stats[C + c] = rsqrtf(var + 1e-5f);
    }
}

__global__ void bn_apply(const float* __restrict__ X, const float* __restrict__ stats,
                         const float* __restrict__ g, const float* __restrict__ b,
                         float* __restrict__ Y, long total, int C)
{
    const long i = (long)blockIdx.x * blockDim.x + threadIdx.x;
    if (i >= total) return;
    const int c = (int)(i % C);
    Y[i] = g[c] * (X[i] - stats[c]) * stats[C + c] + b[c];
}

__global__ void ewadd(const float* __restrict__ a, const float* __restrict__ b,
                      float* __restrict__ c, long n)
{
    const long i = (long)blockIdx.x * blockDim.x + threadIdx.x;
    if (i < n) c[i] = a[i] + b[i];
}

__global__ void zerok(float* __restrict__ a, long n)
{
    const long i = (long)blockIdx.x * blockDim.x + threadIdx.x;
    if (i < n) a[i] = 0.f;
}

// GINE message + segment-sum: aggr[dst] += relu(h[src] + ee)
__global__ __launch_bounds__(256)
void edge_msg_scatter(const float* __restrict__ h, const float* __restrict__ ee,
                      const int* __restrict__ src, const int* __restrict__ dst,
                      float* __restrict__ aggr)
{
    const int e = blockIdx.x;
    const int d = threadIdx.x;
    const int s = src[e], t = dst[e];
    float v = h[(long)s * GHD + d] + ee[(long)e * GHD + d];
    v = v > 0.f ? v : 0.f;
    atomicAdd(&aggr[(long)t * GHD + d], v);
}

// ======================= host orchestration =======================
static void launch_gemm(int act,
                        const float* A, int lda, const float* W, int ldw,
                        const float* bias, const float* res, int ldr,
                        float* C, int ldc, int M, int Nn, int K, hipStream_t s)
{
    dim3 g((Nn + 63) / 64, (M + 127) / 128), b(128);
    if (act) gemm_bias_act<1><<<g, b, 0, s>>>(A, lda, W, ldw, bias, res, ldr, C, ldc, M, Nn, K);
    else     gemm_bias_act<0><<<g, b, 0, s>>>(A, lda, W, ldw, bias, res, ldr, C, ldc, M, Nn, K);
}

extern "C" void kernel_launch(void* const* d_in, const int* in_sizes, int n_in,
                              void* d_out, int out_size, void* d_ws, size_t ws_size,
                              hipStream_t stream)
{
    const float* x         = (const float*)d_in[0];
    const float* pe        = (const float*)d_in[1];
    const float* edge_attr = (const float*)d_in[2];
    const int*   ei        = (const int*)d_in[3];
    const float* np_w1 = (const float*)d_in[4];  const float* np_b1 = (const float*)d_in[5];
    const float* np_w2 = (const float*)d_in[6];  const float* np_b2 = (const float*)d_in[7];
    const float* pebn_g = (const float*)d_in[8]; const float* pebn_b = (const float*)d_in[9];
    const float* pe_w = (const float*)d_in[10];  const float* pe_b = (const float*)d_in[11];
    const float* ep_w1 = (const float*)d_in[12]; const float* ep_b1 = (const float*)d_in[13];
    const float* ep_w2 = (const float*)d_in[14]; const float* ep_b2 = (const float*)d_in[15];
    const float* gine_w1 = (const float*)d_in[16]; const float* gine_b1 = (const float*)d_in[17];
    const float* gine_w2 = (const float*)d_in[18]; const float* gine_b2 = (const float*)d_in[19];
    const float* aq_w = (const float*)d_in[20]; const float* aq_b = (const float*)d_in[21];
    const float* ak_w = (const float*)d_in[22]; const float* ak_b = (const float*)d_in[23];
    const float* av_w = (const float*)d_in[24]; const float* av_b = (const float*)d_in[25];
    const float* ao_w = (const float*)d_in[26]; const float* ao_b = (const float*)d_in[27];
    const float* bn1_g = (const float*)d_in[28]; const float* bn1_b = (const float*)d_in[29];
    const float* bn2_g = (const float*)d_in[30]; const float* bn2_b = (const float*)d_in[31];
    const float* bn3_g = (const float*)d_in[32]; const float* bn3_b = (const float*)d_in[33];
    const float* ff_w1 = (const float*)d_in[34]; const float* ff_b1 = (const float*)d_in[35];
    const float* ff_w2 = (const float*)d_in[36]; const float* ff_b2 = (const float*)d_in[37];

    // workspace carve-up (floats)
    float* wsf = (float*)d_ws;
    size_t o = 0;
    float* h    = wsf + o; o += (size_t)GN * GHD;
    float* bufA = wsf + o; o += (size_t)GN * GHD;
    float* bufB = wsf + o; o += (size_t)GN * GHD;
    float* bufC = wsf + o; o += (size_t)GN * GHD;
    float* bufD = wsf + o; o += (size_t)GN * GHD;
    float* bufQ = wsf + o; o += (size_t)GN * GHD;
    float* bufK = wsf + o; o += (size_t)GN * GHD;
    float* bufV = wsf + o; o += (size_t)GN * GHD;
    float* bufF = wsf + o; o += (size_t)GN * 2 * GHD;
    float* bufP = wsf + o; o += (size_t)GN * GPID;
    float* stats = wsf + o; o += 1024;
    float* ehid = wsf + o; o += (size_t)GE * GHD;
    float* ee   = wsf + o; o += (size_t)GE * GHD;
    (void)ws_size; (void)n_in; (void)in_sizes; (void)out_size;

    const int* src = ei;
    const int* dst = ei + GE;
    const long nhd = (long)GN * GHD;
    const int  ew_blocks = (int)((nhd + 255) / 256);

    // ---- PE branch: BN(pe) @ pe_w + pe_b -> h[:,192:256]
    bn_stats<<<GPID, 256, 0, stream>>>(pe, GN, GPID, stats);
    bn_apply<<<(int)(((long)GN * GPID + 255) / 256), 256, 0, stream>>>(
        pe, stats, pebn_g, pebn_b, bufP, (long)GN * GPID, GPID);
    launch_gemm(0, bufP, GPID, pe_w, 64, pe_b, nullptr, 0, h + GNPD, GHD, GN, 64, GPID, stream);

    // ---- node encoder MLP -> h[:, :192]
    launch_gemm(1, x, 128, np_w1, GNPD, np_b1, nullptr, 0, bufA, GNPD, GN, GNPD, 128, stream);
    launch_gemm(0, bufA, GNPD, np_w2, GNPD, np_b2, nullptr, 0, h, GHD, GN, GNPD, GNPD, stream);

    // ---- edge encoder MLP -> ee [E, 256]
    launch_gemm(1, edge_attr, 64, ep_w1, GHD, ep_b1, nullptr, 0, ehid, GHD, GE, GHD, 64, stream);
    launch_gemm(0, ehid, GHD, ep_w2, GHD, ep_b2, nullptr, 0, ee, GHD, GE, GHD, GHD, stream);

    for (int l = 0; l < 3; ++l) {
        const size_t wo = (size_t)l * GHD * GHD;   // 256x256 weight slab offset
        const size_t bo = (size_t)l * GHD;         // 256 bias offset
        const size_t fo = (size_t)l * GHD * 2 * GHD;

        // GINE: aggr = segsum(relu(h[src]+ee)); t = h + aggr
        zerok<<<ew_blocks, 256, 0, stream>>>(bufA, nhd);
        edge_msg_scatter<<<GE, 256, 0, stream>>>(h, ee, src, dst, bufA);
        ewadd<<<ew_blocks, 256, 0, stream>>>(bufA, h, bufA, nhd);
        launch_gemm(1, bufA, GHD, gine_w1 + wo, GHD, gine_b1 + bo, nullptr, 0, bufB, GHD, GN, GHD, GHD, stream);
        launch_gemm(0, bufB, GHD, gine_w2 + wo, GHD, gine_b2 + bo, h, GHD, bufC, GHD, GN, GHD, GHD, stream);
        bn_stats<<<GHD, 256, 0, stream>>>(bufC, GN, GHD, stats);
        bn_apply<<<ew_blocks, 256, 0, stream>>>(bufC, stats, bn1_g + bo, bn1_b + bo, bufC, nhd, GHD); // hl

        // attention
        launch_gemm(0, h, GHD, aq_w + wo, GHD, aq_b + bo, nullptr, 0, bufQ, GHD, GN, GHD, GHD, stream);
        launch_gemm(0, h, GHD, ak_w + wo, GHD, ak_b + bo, nullptr, 0, bufK, GHD, GN, GHD, GHD, stream);
        launch_gemm(0, h, GHD, av_w + wo, GHD, av_b + bo, nullptr, 0, bufV, GHD, GN, GHD, GHD, stream);
        flash_attn<<<dim3(GN / 64, 4), 128, 0, stream>>>(bufQ, bufK, bufV, bufB);
        launch_gemm(0, bufB, GHD, ao_w + wo, GHD, ao_b + bo, h, GHD, bufD, GHD, GN, GHD, GHD, stream);
        bn_stats<<<GHD, 256, 0, stream>>>(bufD, GN, GHD, stats);
        bn_apply<<<ew_blocks, 256, 0, stream>>>(bufD, stats, bn2_g + bo, bn2_b + bo, bufD, nhd, GHD); // ha

        // out = hl + ha; FFN + residual; BN3 -> h (or d_out on last layer)
        ewadd<<<ew_blocks, 256, 0, stream>>>(bufC, bufD, bufA, nhd);
        launch_gemm(1, bufA, GHD, ff_w1 + fo, 2 * GHD, ff_b1 + (size_t)l * 2 * GHD, nullptr, 0,
                    bufF, 2 * GHD, GN, 2 * GHD, GHD, stream);
        launch_gemm(0, bufF, 2 * GHD, ff_w2 + fo, GHD, ff_b2 + bo, bufA, GHD, bufB, GHD, GN, GHD, 2 * GHD, stream);
        bn_stats<<<GHD, 256, 0, stream>>>(bufB, GN, GHD, stats);
        float* dstp = (l == 2) ? (float*)d_out : h;
        bn_apply<<<ew_blocks, 256, 0, stream>>>(bufB, stats, bn3_g + bo, bn3_b + bo, dstp, nhd, GHD);
    }
}
